// DLEDMD_11081015624082
// MI455X (gfx1250) — matrix operand order repaired and checked by reading, not compiled, not run on hardware
//
#include <hip/hip_runtime.h>

// ---------------------------------------------------------------------------
// DLEDMD for MI455X (gfx1250).
// All MLP GEMMs: v_wmma_f32_16x16x32_bf16, bf16 operands pre-converted once,
// 128-bit fragment loads, 16x64 output strip per wave (4 WMMA per A fetch).
// Shapes: B=128, T=256, D=L=64, H=256, P=4; M = B*T = 32768.
// ---------------------------------------------------------------------------

typedef __attribute__((ext_vector_type(16))) __bf16 v16bf;
typedef __attribute__((ext_vector_type(8)))  __bf16 v8bf;
typedef __attribute__((ext_vector_type(8)))  float  v8f;

__device__ __forceinline__ __bf16 f2bf(float f) {
  union { float f; unsigned u; } in; in.f = f;
  unsigned r = (in.u + 0x7FFFu + ((in.u >> 16) & 1u)) >> 16;  // RNE truncation
  union { unsigned short s; __bf16 b; } out; out.s = (unsigned short)r;
  return out.b;
}

// ---------------------------------------------------------------------------
// Conversion helpers (run once per launch; tiny vs. the GEMM work)
// ---------------------------------------------------------------------------
__global__ void f32_to_bf16(const float* __restrict__ in, __bf16* __restrict__ out, long n) {
  long i = (long)blockIdx.x * blockDim.x + threadIdx.x;
  const long stride = (long)gridDim.x * blockDim.x;
  for (; i < n; i += stride) out[i] = f2bf(in[i]);
}

// W (K x N, f32, row-major) -> Wt (N x K, bf16, row-major)
__global__ void transpose_to_bf16(const float* __restrict__ W, __bf16* __restrict__ Wt,
                                  int K, int N) {
  int idx = blockIdx.x * blockDim.x + threadIdx.x;
  const int total = K * N;
  if (idx >= total) return;
  const int k = idx / N, n = idx - k * N;
  Wt[(size_t)n * K + k] = f2bf(W[idx]);
}

__global__ void fill_zero(float* __restrict__ p, long n) {
  long i = (long)blockIdx.x * blockDim.x + threadIdx.x;
  const long stride = (long)gridDim.x * blockDim.x;
  for (; i < n; i += stride) p[i] = 0.0f;
}

// ---------------------------------------------------------------------------
// WMMA GEMM: C[M x ND] = act(A[M x KD] * W + bias), W given transposed bf16.
// 256 threads = 8 waves; each wave owns a 16x64 output strip (4 subtiles).
// Fragment layouts per CDNA5 ISA 7.12.2 (wave32):
//   A 16x32 bf16: lane half 'hi' holds K runs [hi*8, hi*8+8) and [16+hi*8, ...)
//   B 32x16 bf16: lane half 'hi' holds K runs [hi*8, hi*8+8) and [16+hi*8, ...)
//   C/D f32: row = vgpr + 8*hi, col = lane&15
// ---------------------------------------------------------------------------
template <int KD, int ND, bool RELU, bool WF32, bool WBF16>
__global__ __launch_bounds__(256) void gemm_wmma(const __bf16* __restrict__ A,
                                                 const __bf16* __restrict__ Wt,
                                                 const float* __restrict__ bias,
                                                 float* __restrict__ Cf,
                                                 __bf16* __restrict__ Cb, int M) {
  const int lane = threadIdx.x & 31;
  const int wave = threadIdx.x >> 5;
  const int stripsN = ND / 64;
  const int strip = blockIdx.x * 8 + wave;
  const int totalStrips = (M / 16) * stripsN;
  if (strip >= totalStrips) return;
  const int tm = strip / stripsN;
  const int tn = strip - tm * stripsN;
  const int row0 = tm * 16, col0 = tn * 64;
  const int r  = lane & 15;
  const int hi = lane >> 4;

  v8f acc[4] = {{}, {}, {}, {}};
  const __bf16* Arow = A + (size_t)(row0 + r) * KD + hi * 8;

  for (int k0 = 0; k0 < KD; k0 += 32) {
    if (k0 + 32 < KD)
      __builtin_prefetch(Arow + k0 + 32, 0, 1);        // global_prefetch_b8

    union { v16bf v; v8bf h[2]; } a;
    a.h[0] = *(const v8bf*)(Arow + k0);                // K = k0+hi*8 .. +8
    a.h[1] = *(const v8bf*)(Arow + k0 + 16);           // K = k0+16+hi*8 .. +8

#pragma unroll
    for (int s = 0; s < 4; ++s) {
      const __bf16* bp = Wt + (size_t)(col0 + s * 16 + r) * KD + k0 + hi * 8;
      union { v16bf v; v8bf h[2]; } b;
      b.h[0] = *(const v8bf*)(bp);                     // K = k0+hi*8 .. +8
      b.h[1] = *(const v8bf*)(bp + 16);                // K = k0+16+hi*8 .. +8
      acc[s] = __builtin_amdgcn_wmma_f32_16x16x32_bf16(
          false, a.v, false, b.v, (short)0, acc[s], false, false);
    }
  }

#pragma unroll
  for (int s = 0; s < 4; ++s) {
    const int n = col0 + s * 16 + r;
    const float bv = bias[n];
#pragma unroll
    for (int v = 0; v < 8; ++v) {
      const int m = row0 + v + hi * 8;
      float val = acc[s][v] + bv;
      if (RELU) val = fmaxf(val, 0.0f);
      if (WF32)  Cf[(size_t)m * ND + n] = val;
      if (WBF16) Cb[(size_t)m * ND + n] = f2bf(val);
    }
  }
}

// ---------------------------------------------------------------------------
// EDMD normal equations, per batch b (f32, tiny work):
//   G[i][j]  = sum_{t<255} y[b,t,i]*y[b,t,j];  Ap[i][j] = sum y[b,t+1,i]*y[b,t,j]
// ---------------------------------------------------------------------------
__global__ __launch_bounds__(256) void edmd_gram(const float* __restrict__ y,
                                                 float* __restrict__ G,
                                                 float* __restrict__ ApT) {
  __shared__ float sy[65 * 64];
  const int b = blockIdx.x, tid = threadIdx.x;
  float gacc[16], aacc[16];
#pragma unroll
  for (int p = 0; p < 16; ++p) { gacc[p] = 0.0f; aacc[p] = 0.0f; }

  const float* yb = y + (size_t)b * 256 * 64;
  for (int c = 0; c < 4; ++c) {
    const int t0 = c * 64;
    const int rows = (t0 + 65 <= 256) ? 65 : (256 - t0);
    for (int idx = tid; idx < rows * 64; idx += 256)
      sy[idx] = yb[t0 * 64 + idx];
    __syncthreads();
    const int tmax = (255 - t0 < 64) ? (255 - t0) : 64;
#pragma unroll
    for (int p = 0; p < 16; ++p) {
      const int pair = tid * 16 + p;
      const int i = pair >> 6, j = pair & 63;
      float g = gacc[p], a = aacc[p];
      for (int t = 0; t < tmax; ++t) {
        const float yj = sy[t * 64 + j];
        g += sy[t * 64 + i] * yj;
        a += sy[(t + 1) * 64 + i] * yj;
      }
      gacc[p] = g; aacc[p] = a;
    }
    __syncthreads();
  }
#pragma unroll
  for (int p = 0; p < 16; ++p) {
    const int pair = tid * 16 + p;
    const int i = pair >> 6, j = pair & 63;
    G[(size_t)b * 4096 + pair] = gacc[p];
    ApT[(size_t)b * 4096 + j * 64 + i] = aacc[p];
  }
}

// Solve G X = Ap^T by Gauss-Jordan in LDS, K = X^T.  64 threads per batch.
__global__ __launch_bounds__(64) void edmd_solve(const float* __restrict__ G,
                                                 const float* __restrict__ ApT,
                                                 float* __restrict__ K) {
  __shared__ float A[64 * 64];
  __shared__ float R[64 * 64];
  const int b = blockIdx.x, tid = threadIdx.x;
  for (int r = 0; r < 64; ++r) {
    A[r * 64 + tid] = G[(size_t)b * 4096 + r * 64 + tid];
    R[r * 64 + tid] = ApT[(size_t)b * 4096 + r * 64 + tid];
  }
  __syncthreads();
  for (int p = 0; p < 64; ++p) {
    const float piv = A[p * 64 + p];
    const float inv = 1.0f / (piv + (piv >= 0.0f ? 1e-20f : -1e-20f));
    A[p * 64 + tid] *= inv;
    R[p * 64 + tid] *= inv;
    __syncthreads();
    if (tid != p) {
      const float f = A[tid * 64 + p];
      for (int j = 0; j < 64; ++j) {
        A[tid * 64 + j] -= f * A[p * 64 + j];
        R[tid * 64 + j] -= f * R[p * 64 + j];
      }
    }
    __syncthreads();
  }
  for (int r = 0; r < 64; ++r)
    K[(size_t)b * 4096 + r * 64 + tid] = R[tid * 64 + r];
}

// y_adv_real[b,s,:] = K^{2^{s+1}} @ x[b,T-1,:]
__global__ __launch_bounds__(64) void edmd_advance(const float* __restrict__ x,
                                                   const float* __restrict__ K,
                                                   float* __restrict__ yadv) {
  __shared__ float v[64];
  const int b = blockIdx.x, tid = threadIdx.x;
  v[tid] = x[(size_t)b * 256 * 64 + 255 * 64 + tid];
  __syncthreads();
  const float* Kb = K + (size_t)b * 4096;
  for (int s = 0; s < 4; ++s) {
    const int reps = (s < 2) ? 2 : (s == 2 ? 4 : 8);
    for (int rr = 0; rr < reps; ++rr) {
      float acc = 0.0f;
      for (int j = 0; j < 64; ++j) acc += Kb[tid * 64 + j] * v[j];
      __syncthreads();
      v[tid] = acc;
      __syncthreads();
    }
    yadv[(size_t)b * 256 + s * 64 + tid] = v[tid];
  }
}

// ---------------------------------------------------------------------------
extern "C" void kernel_launch(void* const* d_in, const int* in_sizes, int n_in,
                              void* d_out, int out_size, void* d_ws, size_t ws_size,
                              hipStream_t stream) {
  (void)in_sizes; (void)n_in; (void)ws_size;
  const float* x = (const float*)d_in[0];
  const float *ew[4], *eb[4], *dw[4], *db[4];
  for (int i = 0; i < 4; ++i) {
    ew[i] = (const float*)d_in[1 + 2 * i];
    eb[i] = (const float*)d_in[2 + 2 * i];
    dw[i] = (const float*)d_in[9 + 2 * i];
    db[i] = (const float*)d_in[10 + 2 * i];
  }

  const int M  = 32768;  // B*T
  const int Ma = 512;    // B*P

  // ---- workspace bump allocator (256B aligned) ----
  char* wsb = (char*)d_ws;
  auto alloc = [&](size_t bytes) -> char* {
    char* p = wsb;
    wsb += (bytes + 255) & ~(size_t)255;
    return p;
  };
  __bf16* xbf    = (__bf16*)alloc((size_t)M * 64 * 2);
  __bf16* ab0    = (__bf16*)alloc((size_t)M * 256 * 2);
  __bf16* ab1    = (__bf16*)alloc((size_t)M * 256 * 2);
  __bf16* ybf    = (__bf16*)alloc((size_t)M * 64 * 2);
  __bf16* yadvbf = (__bf16*)alloc((size_t)Ma * 64 * 2);
  static const int KDIM[4] = {64, 256, 256, 256};
  static const int NDIM[4] = {256, 256, 256, 64};
  __bf16 *ewt[4], *dwt[4];
  for (int i = 0; i < 4; ++i) ewt[i] = (__bf16*)alloc((size_t)KDIM[i] * NDIM[i] * 2);
  for (int i = 0; i < 4; ++i) dwt[i] = (__bf16*)alloc((size_t)KDIM[i] * NDIM[i] * 2);
  float* G   = (float*)alloc((size_t)128 * 4096 * 4);
  float* ApT = (float*)alloc((size_t)128 * 4096 * 4);
  float* Km  = (float*)alloc((size_t)128 * 4096 * 4);

  // ---- output layout (f32, flat in return order) ----
  float* out  = (float*)d_out;
  float* y    = out;                    // (B,T,L)
  float* xae  = out + (size_t)2097152;  // (B,T,D)
  float* xadv = out + (size_t)4194304;  // (B,P,D)
  float* yadv = out + (size_t)4227072;  // (B,P,L); imag + eig outputs stay zero

  fill_zero<<<4096, 256, 0, stream>>>(out, (long)out_size);

  // ---- one-time conversions ----
  f32_to_bf16<<<2048, 256, 0, stream>>>(x, xbf, (long)M * 64);
  for (int i = 0; i < 4; ++i) {
    const int tot = KDIM[i] * NDIM[i];
    transpose_to_bf16<<<(tot + 255) / 256, 256, 0, stream>>>(ew[i], ewt[i], KDIM[i], NDIM[i]);
    transpose_to_bf16<<<(tot + 255) / 256, 256, 0, stream>>>(dw[i], dwt[i], KDIM[i], NDIM[i]);
  }

#define NB(MM, NN) ((((MM) / 16) * ((NN) / 64) + 7) / 8)
  // encoder: 64->256->256->256->64 ; last layer writes f32 y + bf16 ybf
  gemm_wmma<64, 256, true, false, true ><<<NB(M, 256), 256, 0, stream>>>(xbf, ewt[0], eb[0], nullptr, ab0, M);
  gemm_wmma<256, 256, true, false, true ><<<NB(M, 256), 256, 0, stream>>>(ab0, ewt[1], eb[1], nullptr, ab1, M);
  gemm_wmma<256, 256, true, false, true ><<<NB(M, 256), 256, 0, stream>>>(ab1, ewt[2], eb[2], nullptr, ab0, M);
  gemm_wmma<256, 64, false, true, true ><<<NB(M, 64),  256, 0, stream>>>(ab0, ewt[3], eb[3], y, ybf, M);

  // decoder (autoencode): 64->256->256->256->64 -> x_ae (f32)
  gemm_wmma<64, 256, true, false, true ><<<NB(M, 256), 256, 0, stream>>>(ybf, dwt[0], db[0], nullptr, ab0, M);
  gemm_wmma<256, 256, true, false, true ><<<NB(M, 256), 256, 0, stream>>>(ab0, dwt[1], db[1], nullptr, ab1, M);
  gemm_wmma<256, 256, true, false, true ><<<NB(M, 256), 256, 0, stream>>>(ab1, dwt[2], db[2], nullptr, ab0, M);
  gemm_wmma<256, 64, false, true, false><<<NB(M, 64),  256, 0, stream>>>(ab0, dwt[3], db[3], xae, nullptr, M);

  // EDMD: normal equations -> K -> rollout K^{2,4,8,16} x_last
  edmd_gram<<<128, 256, 0, stream>>>(y, G, ApT);
  edmd_solve<<<128, 64, 0, stream>>>(G, ApT, Km);
  edmd_advance<<<128, 64, 0, stream>>>(x, Km, yadv);

  // decoder on advanced latents -> x_adv
  f32_to_bf16<<<128, 256, 0, stream>>>(yadv, yadvbf, (long)Ma * 64);
  gemm_wmma<64, 256, true, false, true ><<<NB(Ma, 256), 256, 0, stream>>>(yadvbf, dwt[0], db[0], nullptr, ab0, Ma);
  gemm_wmma<256, 256, true, false, true ><<<NB(Ma, 256), 256, 0, stream>>>(ab0, dwt[1], db[1], nullptr, ab1, Ma);
  gemm_wmma<256, 256, true, false, true ><<<NB(Ma, 256), 256, 0, stream>>>(ab1, dwt[2], db[2], nullptr, ab0, Ma);
  gemm_wmma<256, 64, false, true, false><<<NB(Ma, 64),  256, 0, stream>>>(ab0, dwt[3], db[3], xadv, nullptr, Ma);
#undef NB
}
